// MultiHeadAttention_62517543960997
// MI455X (gfx1250) — matrix operand order repaired and checked by reading, not compiled, not run on hardware
//
#include <hip/hip_runtime.h>

// ---------------------------------------------------------------------------
// MultiHeadAttention forward for MI455X (gfx1250, wave32, WMMA f16->f32)
// n=8 batch, c=2048 seq, d=512 model, 8 heads, hd=64. scale = 8/64 = 0.125
// Pipeline: pack -> QKV GEMMs (WMMA, async-LDS B panel) -> flash attention
//           (WMMA, online softmax) -> output GEMM
// ---------------------------------------------------------------------------

#define NB   8
#define SEQ  2048
#define DM   512
#define NH   8
#define HD   64
#define RTOT (NB * SEQ)   // 16384 total rows

typedef __attribute__((ext_vector_type(16))) _Float16 v16h;
typedef __attribute__((ext_vector_type(8)))  _Float16 v8h;
typedef __attribute__((ext_vector_type(8)))  float    v8f;
typedef int v4i __attribute__((vector_size(16)));

union V16 { v16h v; v8h h[2]; _Float16 e[16]; };

// CDNA5 async global->LDS path (CDNA5 ISA), guarded so the file
// compiles on toolchains without the builtins (falls back to load+ds_store).
#if defined(__HIP_DEVICE_COMPILE__) &&                                        \
    __has_builtin(__builtin_amdgcn_global_load_async_to_lds_b128) &&          \
    __has_builtin(__builtin_amdgcn_s_wait_asynccnt)
#define USE_ASYNC_LDS 1
typedef __attribute__((address_space(1))) v4i gvec_t;
typedef __attribute__((address_space(3))) v4i lvec_t;
#else
#define USE_ASYNC_LDS 0
#endif

// A-matrix (16x32 f16) element K-index for lane-half hi, element e (ISA 7.12.2)
__device__ __forceinline__ int ka(int e, int hi) {
  return (e < 8) ? (hi * 8 + e) : (16 + hi * 8 + (e - 8));
}
// B-matrix (32x16 f16): same K order as A, K = ka(e, hi); column N = lane % 16

__device__ __forceinline__ v8f wmma16(v16h a, v16h b, v8f c) {
  return __builtin_amdgcn_wmma_f32_16x16x32_f16(false, a, false, b, (short)0, c,
                                                false, false);
}

// Copy one lane's 32B of a fragment global -> LDS (async on CDNA5).
__device__ __forceinline__ void copy_frag32B(const _Float16* g, _Float16* l) {
#if USE_ASYNC_LDS
  gvec_t* gp = (gvec_t*)const_cast<_Float16*>(g);
  lvec_t* lp = (lvec_t*)l;
  __builtin_amdgcn_global_load_async_to_lds_b128(gp, lp, 0, 0);
  __builtin_amdgcn_global_load_async_to_lds_b128(gp, lp, 16, 0);
#else
  *(v16h*)l = *(const v16h*)g;
#endif
}

__device__ __forceinline__ void pipe_sync() {
#if USE_ASYNC_LDS
  __builtin_amdgcn_s_wait_asynccnt(0);
#endif
  __syncthreads();
}

// ---------------------------------------------------------------------------
// Pack a row-major fp32 (Mrows x 512) matrix into A-fragment layout (f16).
// Frag f = mt*16 + kt ; per lane 16 halfs (32B) contiguous.
// ---------------------------------------------------------------------------
__global__ void pack_a_kernel(const float* __restrict__ src,
                              _Float16* __restrict__ dst, int Mrows) {
  const int t    = blockIdx.x * blockDim.x + threadIdx.x;
  const int lane = t & 31;
  const int f    = t >> 5;
  const int kt   = f & 15;          // DM/32 = 16 K-tiles
  const int mt   = f >> 4;
  if (mt >= Mrows / 16) return;
  const int hi = lane >> 4, lo = lane & 15;
  const float* s = src + (size_t)(mt * 16 + lo) * DM + kt * 32;
  V16 u;
#pragma unroll
  for (int e = 0; e < 16; ++e) u.e[e] = (_Float16)s[ka(e, hi)];
  *(v16h*)(dst + ((size_t)f * 32 + lane) * 16) = u.v;
}

// ---------------------------------------------------------------------------
// Pack a (512 x 512) weight W (y = x @ W^T) into B-fragment layout:
// B[k][n] = W[n][k].  Frag f = nt*16 + kt.
// ---------------------------------------------------------------------------
__global__ void pack_w_kernel(const float* __restrict__ W,
                              _Float16* __restrict__ dst) {
  const int t    = blockIdx.x * blockDim.x + threadIdx.x;
  const int lane = t & 31;
  const int f    = t >> 5;          // 512 frags (32 nt * 16 kt)
  const int kt   = f & 15;
  const int nt   = f >> 4;
  if (nt >= DM / 16) return;
  const int hi = lane >> 4, lo = lane & 15;
  const float* s = W + (size_t)(nt * 16 + lo) * DM + kt * 32;
  V16 u;
#pragma unroll
  for (int e = 0; e < 16; ++e) u.e[e] = (_Float16)s[ka(e, hi)];
  *(v16h*)(dst + ((size_t)f * 32 + lane) * 16) = u.v;
}

// ---------------------------------------------------------------------------
// GEMM: Y = A(Mrows x 512, A-frags) * B(512 x 512, B-frags) + bias, * outScale
// Block = 4 waves sharing one 64-wide column panel (B staged in LDS via
// async-to-LDS, double buffered, 4x reuse); each wave owns a 64x64 tile.
// Epilogue modes (LDS-transposed where layout requires):
//   0 = row-major fp32 to out (final projection -> d_out)
//   1 = A-fragment f16 (q projection / feeds attention)
//   2 = per-(nb,h) K^T B-fragments (k projection, scale folded in)
//   3 = per-(nb,h) V   B-fragments (v projection)
// ---------------------------------------------------------------------------
__global__ __launch_bounds__(128) void gemm64_kernel(
    const _Float16* __restrict__ A, const _Float16* __restrict__ B,
    const float* __restrict__ bias, void* __restrict__ outp, int Mrows,
    int mode, float outScale) {
  __shared__ __align__(16) _Float16 stage[4][64][72];
  __shared__ __align__(16) _Float16 ldsB[2][4 * 32 * 16];  // 2 x 4KB K-slices
  const int lane = threadIdx.x & 31;
  const int w    = threadIdx.x >> 5;
  const int col0 = (blockIdx.x & 7) * 64;                  // shared by block
  const int row0 = ((blockIdx.x >> 3) * 4 + w) * 64;       // per wave
  (void)Mrows;
  const int hi = lane >> 4, lo = lane & 15;

  // cooperative B K-slice copy: wave w fetches frag (col0/16 + w, kb)
  const size_t bNt = (size_t)(col0 / 16 + w);
  // prologue: kb = 0 into buffer 0
  copy_frag32B(B + ((bNt * 16 + 0) * 32 + lane) * 16,
               &ldsB[0][(w * 32 + lane) * 16]);
  pipe_sync();

  v8f acc[4][4] = {};
  int buf = 0;
#pragma unroll 1
  for (int kb = 0; kb < 16; ++kb) {
    if (kb < 15)
      copy_frag32B(B + ((bNt * 16 + (kb + 1)) * 32 + lane) * 16,
                   &ldsB[buf ^ 1][(w * 32 + lane) * 16]);
    v16h a[4], b[4];
#pragma unroll
    for (int mi = 0; mi < 4; ++mi)
      a[mi] = *(const v16h*)(A + ((((size_t)(row0 / 16 + mi)) * 16 + kb) * 32 + lane) * 16);
#pragma unroll
    for (int ni = 0; ni < 4; ++ni)
      b[ni] = *(const v16h*)&ldsB[buf][(ni * 32 + lane) * 16];
#pragma unroll
    for (int mi = 0; mi < 4; ++mi)
#pragma unroll
      for (int ni = 0; ni < 4; ++ni)
        acc[mi][ni] = wmma16(a[mi], b[ni], acc[mi][ni]);
    pipe_sync();
    buf ^= 1;
  }

  float bv[4];
#pragma unroll
  for (int ni = 0; ni < 4; ++ni) bv[ni] = bias[col0 + ni * 16 + lo];
#pragma unroll
  for (int mi = 0; mi < 4; ++mi)
#pragma unroll
    for (int ni = 0; ni < 4; ++ni)
#pragma unroll
      for (int v = 0; v < 8; ++v)
        acc[mi][ni][v] = (acc[mi][ni][v] + bv[ni]) * outScale;

  if (mode == 0) {  // direct row-major fp32 (C-layout needs no transpose)
    float* o = (float*)outp;
#pragma unroll
    for (int mi = 0; mi < 4; ++mi)
#pragma unroll
      for (int ni = 0; ni < 4; ++ni)
#pragma unroll
        for (int v = 0; v < 8; ++v)
          o[(size_t)(row0 + mi * 16 + v + 8 * hi) * DM + (col0 + ni * 16 + lo)] =
              acc[mi][ni][v];
    return;
  }

  // stage tile in LDS (f16) for layout transpose
#pragma unroll
  for (int mi = 0; mi < 4; ++mi)
#pragma unroll
    for (int ni = 0; ni < 4; ++ni)
#pragma unroll
      for (int v = 0; v < 8; ++v)
        stage[w][mi * 16 + v + 8 * hi][ni * 16 + lo] = (_Float16)acc[mi][ni][v];
  asm volatile("s_wait_dscnt 0" ::: "memory");

  _Float16* o = (_Float16*)outp;
  if (mode == 1) {  // A-fragment layout over (Mrows x 512)
#pragma unroll
    for (int mi = 0; mi < 4; ++mi)
#pragma unroll
      for (int kt2 = 0; kt2 < 2; ++kt2) {
        V16 u;
        u.h[0] = *(const v8h*)&stage[w][mi * 16 + lo][kt2 * 32 + hi * 8];
        u.h[1] = *(const v8h*)&stage[w][mi * 16 + lo][kt2 * 32 + 16 + hi * 8];
        const size_t mt = (size_t)row0 / 16 + mi;
        const size_t kt = col0 / 32 + kt2;
        *(v16h*)(o + ((mt * 16 + kt) * 32 + lane) * 16) = u.v;
      }
  } else if (mode == 2) {  // K^T B-frags: per (nb,h), K=64 head-dim, N=2048 tokens
    const int nb = row0 / SEQ, tok0 = row0 % SEQ, h = col0 / HD;
#pragma unroll
    for (int ntl = 0; ntl < 4; ++ntl)
#pragma unroll
      for (int kt2 = 0; kt2 < 2; ++kt2) {
        V16 u;
#pragma unroll
        for (int e = 0; e < 16; ++e)
          u.e[e] = stage[w][ntl * 16 + lo][kt2 * 32 + ka(e, hi)];
        const size_t nt = tok0 / 16 + ntl;
        *(v16h*)(o + ((((size_t)(nb * NH + h) * 128 + nt) * 2 + kt2) * 32 + lane) * 16) = u.v;
      }
  } else {  // mode 3: V B-frags: per (nb,h), K=2048 tokens, N=64 head-dim
    const int nb = row0 / SEQ, tok0 = row0 % SEQ, h = col0 / HD;
#pragma unroll
    for (int kt2 = 0; kt2 < 2; ++kt2)
#pragma unroll
      for (int ntl = 0; ntl < 4; ++ntl) {
        V16 u;
#pragma unroll
        for (int e = 0; e < 16; ++e)
          u.e[e] = stage[w][kt2 * 32 + ka(e, hi)][ntl * 16 + lo];
        const size_t kt = tok0 / 32 + kt2;
        *(v16h*)(o + ((((size_t)(nb * NH + h) * 64 + kt) * 4 + ntl) * 32 + lane) * 16) = u.v;
      }
  }
}

// ---------------------------------------------------------------------------
// Flash-style attention: one wave owns 32 queries of one (nb, h); streams keys
// in chunks of 64 with online softmax. Emits O as A-frags for the final GEMM.
// ---------------------------------------------------------------------------
__global__ __launch_bounds__(256) void attn_kernel(
    const _Float16* __restrict__ QF, const _Float16* __restrict__ KF,
    const _Float16* __restrict__ VF, const float* __restrict__ mask,
    _Float16* __restrict__ OF) {
  __shared__ __align__(16) _Float16 pst[8][32][72];
  const int lane = threadIdx.x & 31;
  const int w    = threadIdx.x >> 5;
  const int nb   = blockIdx.x >> 3;
  const int h    = blockIdx.x & 7;
  const int tok0 = blockIdx.y * 256 + w * 32;
  const int hi = lane >> 4, lo = lane & 15;
  const size_t bh = (size_t)(nb * NH + h);

  v16h qf[2][2];
#pragma unroll
  for (int mi = 0; mi < 2; ++mi)
#pragma unroll
    for (int kt2 = 0; kt2 < 2; ++kt2) {
      const size_t mt = (size_t)(nb * SEQ + tok0) / 16 + mi;
      const size_t kt = 2 * h + kt2;
      qf[mi][kt2] = *(const v16h*)(QF + ((mt * 16 + kt) * 32 + lane) * 16);
    }

  float mrun[2][8], lrun[2][8];
#pragma unroll
  for (int mi = 0; mi < 2; ++mi)
#pragma unroll
    for (int v = 0; v < 8; ++v) { mrun[mi][v] = -3.0e38f; lrun[mi][v] = 0.0f; }
  v8f accO[2][4] = {};

  const float* mrow = mask + (size_t)nb * SEQ * SEQ;

#pragma unroll 1
  for (int j = 0; j < SEQ / 64; ++j) {
    if (j + 1 < SEQ / 64) {  // prefetch next chunk's K/V frags (global_prefetch)
      __builtin_prefetch(KF + ((bh * 128 + (size_t)(j + 1) * 4) * 2 * 32) * 16, 0, 1);
      __builtin_prefetch(VF + ((bh * 64 + (size_t)(j + 1) * 2) * 4 * 32) * 16, 0, 1);
    }
    // ---- S = q @ k^T (scale already folded into KF) ----
    v8f s[2][4] = {};
#pragma unroll
    for (int kt2 = 0; kt2 < 2; ++kt2) {
      v16h kf[4];
#pragma unroll
      for (int ni = 0; ni < 4; ++ni)
        kf[ni] = *(const v16h*)(KF + (((bh * 128 + (j * 4 + ni)) * 2 + kt2) * 32 + lane) * 16);
#pragma unroll
      for (int mi = 0; mi < 2; ++mi)
#pragma unroll
        for (int ni = 0; ni < 4; ++ni)
          s[mi][ni] = wmma16(qf[mi][kt2], kf[ni], s[mi][ni]);
    }
    // ---- mask add + online softmax (row stats across 16-lane halves) ----
#pragma unroll
    for (int mi = 0; mi < 2; ++mi) {
#pragma unroll
      for (int v = 0; v < 8; ++v) {
        const int row = tok0 + mi * 16 + v + 8 * hi;
        float cm = -3.0e38f;
#pragma unroll
        for (int ni = 0; ni < 4; ++ni) {
          float x = s[mi][ni][v] + mrow[(size_t)row * SEQ + j * 64 + ni * 16 + lo];
          s[mi][ni][v] = x;
          cm = fmaxf(cm, x);
        }
        cm = fmaxf(cm, __shfl_xor(cm, 1));
        cm = fmaxf(cm, __shfl_xor(cm, 2));
        cm = fmaxf(cm, __shfl_xor(cm, 4));
        cm = fmaxf(cm, __shfl_xor(cm, 8));
        const float mn  = fmaxf(mrun[mi][v], cm);
        const float fsc = __expf(mrun[mi][v] - mn);
        float rs = 0.0f;
#pragma unroll
        for (int ni = 0; ni < 4; ++ni) {
          float p = __expf(s[mi][ni][v] - mn);
          s[mi][ni][v] = p;
          rs += p;
        }
        rs += __shfl_xor(rs, 1);
        rs += __shfl_xor(rs, 2);
        rs += __shfl_xor(rs, 4);
        rs += __shfl_xor(rs, 8);
        lrun[mi][v] = lrun[mi][v] * fsc + rs;
        mrun[mi][v] = mn;
#pragma unroll
        for (int ni = 0; ni < 4; ++ni) {
          accO[mi][ni][v] *= fsc;
          pst[w][mi * 16 + v + 8 * hi][ni * 16 + lo] = (_Float16)s[mi][ni][v];
        }
      }
    }
    asm volatile("s_wait_dscnt 0" ::: "memory");
    // ---- O += P @ V (P read back from LDS in A-fragment layout) ----
#pragma unroll
    for (int kt2 = 0; kt2 < 2; ++kt2) {
      V16 pf[2];
#pragma unroll
      for (int mi = 0; mi < 2; ++mi) {
        pf[mi].h[0] = *(const v8h*)&pst[w][mi * 16 + lo][kt2 * 32 + hi * 8];
        pf[mi].h[1] = *(const v8h*)&pst[w][mi * 16 + lo][kt2 * 32 + 16 + hi * 8];
      }
#pragma unroll
      for (int ni = 0; ni < 4; ++ni) {
        v16h vfr = *(const v16h*)(VF + (((bh * 64 + (j * 2 + kt2)) * 4 + ni) * 32 + lane) * 16);
#pragma unroll
        for (int mi = 0; mi < 2; ++mi)
          accO[mi][ni] = wmma16(pf[mi].v, vfr, accO[mi][ni]);
      }
    }
    asm volatile("s_wait_dscnt 0" ::: "memory");
  }

  // ---- normalize + emit O as A-frags into (16384 x 512) frag buffer ----
#pragma unroll
  for (int mi = 0; mi < 2; ++mi)
#pragma unroll
    for (int v = 0; v < 8; ++v) {
      const float inv = 1.0f / lrun[mi][v];
#pragma unroll
      for (int ni = 0; ni < 4; ++ni)
        pst[w][mi * 16 + v + 8 * hi][ni * 16 + lo] = (_Float16)(accO[mi][ni][v] * inv);
    }
  asm volatile("s_wait_dscnt 0" ::: "memory");
#pragma unroll
  for (int mi = 0; mi < 2; ++mi)
#pragma unroll
    for (int kt2 = 0; kt2 < 2; ++kt2) {
      V16 u;
      u.h[0] = *(const v8h*)&pst[w][mi * 16 + lo][kt2 * 32 + hi * 8];
      u.h[1] = *(const v8h*)&pst[w][mi * 16 + lo][kt2 * 32 + 16 + hi * 8];
      const size_t mt = (size_t)(nb * SEQ + tok0) / 16 + mi;
      const size_t kt = 2 * h + kt2;
      *(v16h*)(OF + ((mt * 16 + kt) * 32 + lane) * 16) = u.v;
    }
}

// ---------------------------------------------------------------------------
extern "C" void kernel_launch(void* const* d_in, const int* in_sizes, int n_in,
                              void* d_out, int out_size, void* d_ws,
                              size_t ws_size, hipStream_t stream) {
  (void)in_sizes; (void)n_in; (void)out_size;
  const float* q_in = (const float*)d_in[0];
  const float* k_in = (const float*)d_in[1];
  const float* v_in = (const float*)d_in[2];
  const float* mask = (const float*)d_in[3];
  const float* Wq = (const float*)d_in[4];  const float* bq = (const float*)d_in[5];
  const float* Wk = (const float*)d_in[6];  const float* bk = (const float*)d_in[7];
  const float* Wv = (const float*)d_in[8];  const float* bv = (const float*)d_in[9];
  const float* Wo = (const float*)d_in[10]; const float* bo = (const float*)d_in[11];

  char* ws = (char*)d_ws;
  const size_t WFRAG_BYTES = (size_t)DM * DM * sizeof(_Float16);   // 512 KiB
  const size_t XFRAG_BYTES = (size_t)RTOT * DM * sizeof(_Float16); // 16 MiB
  if (ws_size < 4 * WFRAG_BYTES + 4 * XFRAG_BYTES) return;

  _Float16* wqf = (_Float16*)(ws + 0 * WFRAG_BYTES);
  _Float16* wkf = (_Float16*)(ws + 1 * WFRAG_BYTES);
  _Float16* wvf = (_Float16*)(ws + 2 * WFRAG_BYTES);
  _Float16* wof = (_Float16*)(ws + 3 * WFRAG_BYTES);
  char* base = ws + 4 * WFRAG_BYTES;
  _Float16* inf = (_Float16*)(base + 0 * XFRAG_BYTES);  // input frags / O frags
  _Float16* qf  = (_Float16*)(base + 1 * XFRAG_BYTES);
  _Float16* kf  = (_Float16*)(base + 2 * XFRAG_BYTES);
  _Float16* vf  = (_Float16*)(base + 3 * XFRAG_BYTES);

  // pack weights into B-fragment layout
  pack_w_kernel<<<64, 256, 0, stream>>>(Wq, wqf);
  pack_w_kernel<<<64, 256, 0, stream>>>(Wk, wkf);
  pack_w_kernel<<<64, 256, 0, stream>>>(Wv, wvf);
  pack_w_kernel<<<64, 256, 0, stream>>>(Wo, wof);

  // q/k/v projections (k gets ATTN_MULT/hd = 0.125 folded into its frags)
  pack_a_kernel<<<2048, 256, 0, stream>>>(q_in, inf, RTOT);
  gemm64_kernel<<<512, 128, 0, stream>>>(inf, wqf, bq, (void*)qf, RTOT, 1, 1.0f);
  pack_a_kernel<<<2048, 256, 0, stream>>>(k_in, inf, RTOT);
  gemm64_kernel<<<512, 128, 0, stream>>>(inf, wkf, bk, (void*)kf, RTOT, 2, 0.125f);
  pack_a_kernel<<<2048, 256, 0, stream>>>(v_in, inf, RTOT);
  gemm64_kernel<<<512, 128, 0, stream>>>(inf, wvf, bv, (void*)vf, RTOT, 3, 1.0f);

  // attention (writes O A-frags into inf), then output projection to d_out
  attn_kernel<<<dim3(64, 8), 256, 0, stream>>>(qf, kf, vf, mask, inf);
  gemm64_kernel<<<512, 128, 0, stream>>>(inf, wof, bo, d_out, RTOT, 0, 1.0f);
}
